// AttentionDecoderModel_82042465288697
// MI455X (gfx1250) — matrix operand order repaired and checked by reading, not compiled, not run on hardware
//
#include <hip/hip_runtime.h>
#include <hip/hip_bf16.h>
#include <math.h>

typedef __attribute__((ext_vector_type(16))) _Float16 v16h;
typedef __attribute__((ext_vector_type(8)))  float    v8f;

#define WMMA_F16(a, b, c) \
  __builtin_amdgcn_wmma_f32_16x16x32_f16(false, (a), false, (b), (short)0, (c), false, false)

namespace cfg {
constexpr int Bv = 32, Sv = 512, Tv = 128, Dv = 512, Av = 512, Hv = 8, HDv = 64;
constexpr int FFv = 2048, Lv = 6, Vv = 500, MDv = 512;
}

// ---------------------------------------------------------------- cvt f32->f16
__global__ void cvt_f32_to_f16(const float* __restrict__ in, _Float16* __restrict__ out, int n) {
  int i = blockIdx.x * blockDim.x + threadIdx.x;
  int stride = gridDim.x * blockDim.x;
  for (; i < n; i += stride) out[i] = (_Float16)in[i];
}

// ------------------------------------------------- embedding + positional enc
__global__ void embed_pe(const int* __restrict__ ids, const float* __restrict__ emb,
                         float* __restrict__ x) {
  const int bt = blockIdx.x;            // 0..B*T-1
  const int t  = bt % cfg::Tv;
  const int id = ids[bt];
  const float sqrtD = 22.627416997969522f;  // sqrt(512)
  for (int d = threadIdx.x; d < cfg::Dv; d += blockDim.x) {
    const int i2 = (d >> 1) * 2;
    const float freq = __expf(-(float)i2 * (9.210340371976184f / (float)cfg::Dv));
    const float ang  = (float)t * freq;
    const float pe   = (d & 1) ? __cosf(ang) : __sinf(ang);
    x[(size_t)bt * cfg::Dv + d] = emb[(size_t)id * cfg::Dv + d] * sqrtD + pe;
  }
}

// -------------------------------------------------------- layernorm -> f16 out
__global__ __launch_bounds__(256) void layernorm_f16(const float* __restrict__ x,
                                                     const float* __restrict__ g,
                                                     const float* __restrict__ b,
                                                     _Float16* __restrict__ out) {
  const int row = blockIdx.x;
  const int tid = threadIdx.x, lane = tid & 31, wv = tid >> 5;
  const float* xr = x + (size_t)row * cfg::Dv;
  const float v0 = xr[tid], v1 = xr[tid + 256];
  __shared__ float red[8];

  float s = v0 + v1;
  for (int m = 16; m; m >>= 1) s += __shfl_xor(s, m, 32);
  if (lane == 0) red[wv] = s;
  __syncthreads();
  float tot = 0.f;
#pragma unroll
  for (int i = 0; i < 8; ++i) tot += red[i];
  const float mean = tot * (1.0f / cfg::Dv);
  __syncthreads();

  const float d0 = v0 - mean, d1 = v1 - mean;
  float q = d0 * d0 + d1 * d1;
  for (int m = 16; m; m >>= 1) q += __shfl_xor(q, m, 32);
  if (lane == 0) red[wv] = q;
  __syncthreads();
  float vq = 0.f;
#pragma unroll
  for (int i = 0; i < 8; ++i) vq += red[i];
  const float rstd = rsqrtf(vq * (1.0f / cfg::Dv) + 1e-5f);

  out[(size_t)row * cfg::Dv + tid]       = (_Float16)(d0 * rstd * g[tid] + b[tid]);
  out[(size_t)row * cfg::Dv + tid + 256] = (_Float16)(d1 * rstd * g[tid + 256] + b[tid + 256]);
}

// ------------------------------------------------------------- WMMA fragments
// A tile in LDS row-major [128][40]; fragment = 2x ds_load_b128 per half-K
__device__ __forceinline__ v16h load_a_lds(_Float16 (*As)[40], int mrow, int lane) {
  v16h a;
  const int m  = mrow + (lane & 15);
  const int kb = (lane >> 4) * 16;
#pragma unroll
  for (int i = 0; i < 8; ++i) {
    a[i]     = As[m][kb + i];
    a[8 + i] = As[m][kb + 8 + i];
  }
  return a;
}
// B tile stored TRANSPOSED in LDS [64 cols][40]; per-lane 16 contiguous halfs
// (row stride 80B -> 16B aligned bases -> two ds_load_b128, no v_mov_b16 packs)
__device__ __forceinline__ v16h load_bt_lds(_Float16 (*Bt)[40], int ncol, int lane) {
  v16h bf;
  const int nn = ncol + (lane & 15);
  const int ko = (lane >> 4) * 16;
#pragma unroll
  for (int i = 0; i < 16; ++i) bf[i] = Bt[nn][ko + i];
  return bf;
}

// -------------------------------------------------- generic fused WMMA GEMM
// C[M,N] = A[M,K](f16) * B[K,N](f16) + bias; optional +resid, Swish,
// f32 out, f16 out, and/or TRANSPOSED f16 out (outhT[ n*ldT + m ], for V^T)
__global__ __launch_bounds__(256)
void wmma_gemm(const _Float16* __restrict__ Ag, const _Float16* __restrict__ Bg,
               const float* __restrict__ bias, const float* __restrict__ resid,
               float* __restrict__ outf, _Float16* __restrict__ outh,
               _Float16* __restrict__ outhT, int ldT,
               int M, int N, int K, int act) {
  __shared__ _Float16 As[128][40];   // 128x32 tile (row-major), +8 pad
  __shared__ _Float16 Bt[64][40];    // 32x64 tile stored transposed, +8 pad
  const int tid = threadIdx.x, lane = tid & 31, wv = tid >> 5;
  const int wm = wv & 3, wn = wv >> 2;   // 4x2 wave grid
  const int mbase = blockIdx.x * 128;
  const int nbase = blockIdx.y * 64;
  const bool nvec = ((nbase + 64) <= N) && ((N & 7) == 0);

  v8f c00 = {}, c01 = {}, c10 = {}, c11 = {};

  for (int k0 = 0; k0 < K; k0 += 32) {
    // ---- stage A tile: 128x32 halfs = 512 chunks of 8
    for (int ch = tid; ch < 512; ch += 256) {
      const int r = ch >> 2, co = (ch & 3) * 8;
      const uint4 v = *(const uint4*)(Ag + (size_t)(mbase + r) * K + k0 + co);
      *(uint2*)&As[r][co]     = make_uint2(v.x, v.y);
      *(uint2*)&As[r][co + 4] = make_uint2(v.z, v.w);
    }
    // ---- stage B tile transposed: thread owns row k0+r, cols co..co+7
    {
      const int r = tid >> 3, co = (tid & 7) * 8;
      if (nvec) {
        const uint4 v = *(const uint4*)(Bg + (size_t)(k0 + r) * N + nbase + co);
        union { uint4 u; _Float16 h[8]; } cvtu;
        cvtu.u = v;
#pragma unroll
        for (int i = 0; i < 8; ++i) Bt[co + i][r] = cvtu.h[i];
      } else {
#pragma unroll
        for (int i = 0; i < 8; ++i) {
          const int col = nbase + co + i;
          Bt[co + i][r] = (col < N) ? Bg[(size_t)(k0 + r) * N + col] : (_Float16)0.f;
        }
      }
    }
    if (k0 + 32 < K) {  // prefetch next K tiles -> global_prefetch_b8
      __builtin_prefetch(Ag + (size_t)(mbase + (tid >> 1)) * K + k0 + 32, 0, 1);
      __builtin_prefetch(Bg + (size_t)(k0 + 32 + (tid >> 3)) * N + nbase, 0, 1);
    }
    __syncthreads();

    const v16h a0 = load_a_lds(As, wm * 32, lane);
    const v16h a1 = load_a_lds(As, wm * 32 + 16, lane);
    const v16h b0 = load_bt_lds(Bt, wn * 32, lane);
    const v16h b1 = load_bt_lds(Bt, wn * 32 + 16, lane);
    c00 = WMMA_F16(a0, b0, c00);
    c01 = WMMA_F16(a0, b1, c01);
    c10 = WMMA_F16(a1, b0, c10);
    c11 = WMMA_F16(a1, b1, c11);
    __syncthreads();
  }

  // ---- epilogue: bias (+resid) (+swish) -> f32 / f16 / f16-transposed
  const int rl = 8 * (lane >> 4);
  const int nl = lane & 15;
  v8f acc[2][2] = {{c00, c01}, {c10, c11}};
#pragma unroll
  for (int mi = 0; mi < 2; ++mi)
#pragma unroll
    for (int ni = 0; ni < 2; ++ni) {
      const int gn = nbase + wn * 32 + ni * 16 + nl;
      if (gn >= N) continue;
      const float bb = bias[gn];
#pragma unroll
      for (int r = 0; r < 8; ++r) {
        const int gm = mbase + wm * 32 + mi * 16 + r + rl;
        float val = acc[mi][ni][r] + bb;
        if (resid) val += resid[(size_t)gm * N + gn];
        if (act == 1) val = val * (1.0f / (1.0f + __expf(-val)));  // Swish
        if (outf)  outf[(size_t)gm * N + gn] = val;
        if (outh)  outh[(size_t)gm * N + gn] = (_Float16)val;
        if (outhT) outhT[(size_t)gn * ldT + gm] = (_Float16)val;
      }
    }
}

// ------------------------------------- fused attention: scores/softmax/P.V
// vth is V TRANSPOSED: vth[(h*HD + col) * ldv + (b*SKV + s)]  (contig in s)
template <int SKV, bool CAUSAL>
__global__ __launch_bounds__(256)
void attn_kernel(const _Float16* __restrict__ qh, const _Float16* __restrict__ kh,
                 const _Float16* __restrict__ vth, int ldv,
                 const int* __restrict__ lens, _Float16* __restrict__ outh) {
  __shared__ float    sc[16][SKV];
  __shared__ _Float16 P[16][SKV];
  const int mtile = blockIdx.x;   // T/16
  const int h     = blockIdx.y;   // head
  const int b     = blockIdx.z;   // batch
  const int tid = threadIdx.x, lane = tid & 31, wv = tid >> 5;
  const int m = lane & 15, kb = (lane >> 4) * 8;
  const int n = lane & 15, ko = (lane >> 4) * 16;

  // Q fragments (16x64 -> two 16x32 chunks), held in registers
  const size_t qrow = (((size_t)b * cfg::Tv) + mtile * 16 + m) * cfg::Av + h * cfg::HDv;
  v16h aq0, aq1;
#pragma unroll
  for (int i = 0; i < 8; ++i) {
    aq0[i]     = qh[qrow + ko + i];
    aq0[8 + i] = qh[qrow + ko + 8 + i];
    aq1[i]     = qh[qrow + 32 + ko + i];
    aq1[8 + i] = qh[qrow + 32 + ko + 8 + i];
  }

  // ---- stage 1: scores S = Q K^T / 8 (skip strictly-future tiles if causal)
  constexpr int CT = SKV / 16;
  const int ctmax = CAUSAL ? (mtile + 1) : CT;
  for (int ct = wv; ct < ctmax; ct += 8) {
    const size_t krow = (((size_t)b * SKV) + ct * 16 + n) * cfg::Av + h * cfg::HDv;
    v16h bk0, bk1;
#pragma unroll
    for (int i = 0; i < 16; ++i) {
      bk0[i] = kh[krow + ko + i];        // contiguous -> global_load_b128 x2
      bk1[i] = kh[krow + 32 + ko + i];
    }
    v8f c = {};
    c = WMMA_F16(aq0, bk0, c);
    c = WMMA_F16(aq1, bk1, c);
    const int rl = 8 * (lane >> 4);
#pragma unroll
    for (int r = 0; r < 8; ++r) sc[r + rl][ct * 16 + n] = c[r] * 0.125f;
  }
  __syncthreads();

  // ---- stage 2: masked softmax, rows 2*wv and 2*wv+1 per wave
  const int len = lens[b];
#pragma unroll
  for (int rr = 0; rr < 2; ++rr) {
    const int row = wv * 2 + rr;
    const int tg  = mtile * 16 + row;
    float mx = -3.0e38f;
    for (int j = lane; j < SKV; j += 32) {
      const bool ok = CAUSAL ? ((j <= tg) && (j < len)) : (j < len);
      mx = fmaxf(mx, ok ? sc[row][j] : -3.0e38f);
    }
    for (int msk = 16; msk; msk >>= 1) mx = fmaxf(mx, __shfl_xor(mx, msk, 32));
    float sum = 0.f;
    for (int j = lane; j < SKV; j += 32) {
      const bool ok = CAUSAL ? ((j <= tg) && (j < len)) : (j < len);
      const float e = ok ? __expf(sc[row][j] - mx) : 0.f;
      sum += e;
      sc[row][j] = e;
    }
    for (int msk = 16; msk; msk >>= 1) sum += __shfl_xor(sum, msk, 32);
    const float inv = (sum > 0.f) ? 1.0f / sum : 0.f;
    for (int j = lane; j < SKV; j += 32) P[row][j] = (_Float16)(sc[row][j] * inv);
  }
  __syncthreads();

  // ---- stage 3: O = P @ V ; 4 waves, one 16x16 output tile each (64 cols)
  if (wv < 4) {
    const _Float16* vcol = vth + (size_t)(h * cfg::HDv + wv * 16 + n) * ldv + (size_t)b * SKV;
    v8f c = {};
    for (int s0 = 0; s0 < SKV; s0 += 32) {
      v16h ap, bv;
#pragma unroll
      for (int i = 0; i < 8; ++i) {
        ap[i]     = P[m][s0 + ko + i];       // contiguous -> ds_load_b128
        ap[8 + i] = P[m][s0 + ko + 8 + i];
      }
#pragma unroll
      for (int i = 0; i < 16; ++i)
        bv[i] = vcol[s0 + ko + i];           // contiguous -> global_load_b128 x2
      c = WMMA_F16(ap, bv, c);
    }
    const int rl = 8 * (lane >> 4);
#pragma unroll
    for (int r = 0; r < 8; ++r)
      outh[(((size_t)b * cfg::Tv) + mtile * 16 + r + rl) * cfg::Av + h * cfg::HDv + wv * 16 + n] =
          (_Float16)c[r];
  }
}

// ============================================================== host driver
extern "C" void kernel_launch(void* const* d_in, const int* in_sizes, int n_in,
                              void* d_out, int out_size, void* d_ws, size_t ws_size,
                              hipStream_t stream) {
  using namespace cfg;
  (void)in_sizes; (void)n_in; (void)out_size; (void)ws_size;

  const float* memory      = (const float*)d_in[0];
  const int*   memory_lens = (const int*)d_in[1];
  const int*   ys_in_pad   = (const int*)d_in[2];
  const int*   ys_in_lens  = (const int*)d_in[3];
  const float* emb         = (const float*)d_in[4];
  const float* w_out       = (const float*)d_in[5];
  const float* b_out       = (const float*)d_in[6];
  const float* ln_sa_g = (const float*)d_in[7];
  const float* ln_sa_b = (const float*)d_in[8];
  const float* sa_wq = (const float*)d_in[9];  const float* sa_bq = (const float*)d_in[10];
  const float* sa_wk = (const float*)d_in[11]; const float* sa_bk = (const float*)d_in[12];
  const float* sa_wv = (const float*)d_in[13]; const float* sa_bv = (const float*)d_in[14];
  const float* sa_wo = (const float*)d_in[15]; const float* sa_bo = (const float*)d_in[16];
  const float* ln_ca_g = (const float*)d_in[17];
  const float* ln_ca_b = (const float*)d_in[18];
  const float* ca_wq = (const float*)d_in[19]; const float* ca_bq = (const float*)d_in[20];
  const float* ca_wk = (const float*)d_in[21]; const float* ca_bk = (const float*)d_in[22];
  const float* ca_wv = (const float*)d_in[23]; const float* ca_bv = (const float*)d_in[24];
  const float* ca_wo = (const float*)d_in[25]; const float* ca_bo = (const float*)d_in[26];
  const float* ln_ff_g = (const float*)d_in[27];
  const float* ln_ff_b = (const float*)d_in[28];
  const float* ff_w1 = (const float*)d_in[29]; const float* ff_b1 = (const float*)d_in[30];
  const float* ff_w2 = (const float*)d_in[31]; const float* ff_b2 = (const float*)d_in[32];
  float* logits = (float*)d_out;

  // ---- workspace carve-up (256B aligned)
  char* wsb = (char*)d_ws;
  size_t off = 0;
  auto alloc = [&](size_t bytes) -> void* {
    void* p = wsb + off;
    off = (off + bytes + 255) & ~(size_t)255;
    return p;
  };
  const int BT = Bv * Tv;      // 4096
  const int BS = Bv * Sv;      // 16384
  float*     x        = (float*)alloc((size_t)BT * Dv * 4);
  _Float16*  tn_h     = (_Float16*)alloc((size_t)BT * Dv * 2);
  _Float16*  q_h      = (_Float16*)alloc((size_t)BT * Av * 2);
  _Float16*  k_h      = (_Float16*)alloc((size_t)BT * Av * 2);
  _Float16*  vt_h     = (_Float16*)alloc((size_t)Av * BT * 2);  // V^T [A][BT]
  _Float16*  attn_h   = (_Float16*)alloc((size_t)BT * Av * 2);
  _Float16*  ffh_h    = (_Float16*)alloc((size_t)BT * FFv * 2);
  _Float16*  xh       = (_Float16*)alloc((size_t)BT * Dv * 2);
  _Float16*  memory_h = (_Float16*)alloc((size_t)BS * MDv * 2);
  _Float16*  kc_h     = (_Float16*)alloc((size_t)BS * Av * 2);
  _Float16*  vct_h    = (_Float16*)alloc((size_t)Av * BS * 2);  // Vc^T [A][BS]
  _Float16*  wbuf     = (_Float16*)alloc((size_t)2 * Dv * FFv * 2); // reused
  _Float16*  wout_h   = (_Float16*)alloc((size_t)Dv * Vv * 2 + 256);
  _Float16* w0 = wbuf;
  _Float16* w1 = wbuf + (size_t)Dv * Av;
  _Float16* w2 = wbuf + (size_t)2 * Dv * Av;
  _Float16* wf2 = wbuf + (size_t)Dv * FFv;

  auto cvt = [&](const float* in, _Float16* outp, int n) {
    cvt_f32_to_f16<<<(n + 255) / 256, 256, 0, stream>>>(in, outp, n);
  };
  auto gemm = [&](const _Float16* A, const _Float16* Bm, const float* bias,
                  const float* resid, float* of, _Float16* oh, _Float16* ohT, int ldT,
                  int M, int N, int K, int act) {
    dim3 g(M / 128, (N + 63) / 64);
    wmma_gemm<<<g, 256, 0, stream>>>(A, Bm, bias, resid, of, oh, ohT, ldT, M, N, K, act);
  };

  // ---- prologue
  embed_pe<<<BT, 256, 0, stream>>>(ys_in_pad, emb, x);
  cvt(memory, memory_h, BS * MDv);

  const dim3 agrid(Tv / 16, Hv, Bv);
  for (int l = 0; l < Lv; ++l) {
    // ===== self-attention =====
    layernorm_f16<<<BT, 256, 0, stream>>>(x, ln_sa_g + (size_t)l * Dv, ln_sa_b + (size_t)l * Dv, tn_h);
    cvt(sa_wq + (size_t)l * Dv * Av, w0, Dv * Av);
    cvt(sa_wk + (size_t)l * Dv * Av, w1, Dv * Av);
    cvt(sa_wv + (size_t)l * Dv * Av, w2, Dv * Av);
    gemm(tn_h, w0, sa_bq + (size_t)l * Av, nullptr, nullptr, q_h, nullptr, 0, BT, Av, Dv, 0);
    gemm(tn_h, w1, sa_bk + (size_t)l * Av, nullptr, nullptr, k_h, nullptr, 0, BT, Av, Dv, 0);
    gemm(tn_h, w2, sa_bv + (size_t)l * Av, nullptr, nullptr, nullptr, vt_h, BT, BT, Av, Dv, 0);
    attn_kernel<128, true><<<agrid, 256, 0, stream>>>(q_h, k_h, vt_h, BT, ys_in_lens, attn_h);
    cvt(sa_wo + (size_t)l * Av * Dv, w0, Av * Dv);
    gemm(attn_h, w0, sa_bo + (size_t)l * Dv, x, x, nullptr, nullptr, 0, BT, Dv, Av, 0);

    // ===== cross-attention =====
    layernorm_f16<<<BT, 256, 0, stream>>>(x, ln_ca_g + (size_t)l * Dv, ln_ca_b + (size_t)l * Dv, tn_h);
    cvt(ca_wq + (size_t)l * Dv * Av, w0, Dv * Av);
    cvt(ca_wk + (size_t)l * MDv * Av, w1, MDv * Av);
    cvt(ca_wv + (size_t)l * MDv * Av, w2, MDv * Av);
    gemm(tn_h, w0, ca_bq + (size_t)l * Av, nullptr, nullptr, q_h, nullptr, 0, BT, Av, Dv, 0);
    gemm(memory_h, w1, ca_bk + (size_t)l * Av, nullptr, nullptr, kc_h, nullptr, 0, BS, Av, MDv, 0);
    gemm(memory_h, w2, ca_bv + (size_t)l * Av, nullptr, nullptr, nullptr, vct_h, BS, BS, Av, MDv, 0);
    attn_kernel<512, false><<<agrid, 256, 0, stream>>>(q_h, kc_h, vct_h, BS, memory_lens, attn_h);
    cvt(ca_wo + (size_t)l * Av * Dv, w0, Av * Dv);
    gemm(attn_h, w0, ca_bo + (size_t)l * Dv, x, x, nullptr, nullptr, 0, BT, Dv, Av, 0);

    // ===== feed-forward (Swish) =====
    layernorm_f16<<<BT, 256, 0, stream>>>(x, ln_ff_g + (size_t)l * Dv, ln_ff_b + (size_t)l * Dv, tn_h);
    cvt(ff_w1 + (size_t)l * Dv * FFv, w0, Dv * FFv);
    gemm(tn_h, w0, ff_b1 + (size_t)l * FFv, nullptr, nullptr, ffh_h, nullptr, 0, BT, FFv, Dv, 1);
    cvt(ff_w2 + (size_t)l * FFv * Dv, wf2, FFv * Dv);
    gemm(ffh_h, wf2, ff_b2 + (size_t)l * Dv, x, x, nullptr, nullptr, 0, BT, Dv, FFv, 0);
  }

  // ---- logits: x @ w_out + b_out  (N=500, guarded tiles)
  cvt(x, xh, BT * Dv);
  cvt(w_out, wout_h, Dv * Vv);
  gemm(xh, wout_h, b_out, nullptr, logits, nullptr, nullptr, 0, BT, Vv, Dv, 0);
}
